// BasicRGCN_74947179315434
// MI455X (gfx1250) — matrix operand order repaired and checked by reading, not compiled, not run on hardware
//
#include <hip/hip_runtime.h>
#include <hip/hip_bf16.h>

// R=8, N=4096, F=256, B=16384
#define RNUM 8
#define NN   4096
#define FF   256
#define BB   16384
#define LPITCH 40   // LDS row pitch in bf16 elements (80B, 16B-aligned)

typedef __bf16 v16bf __attribute__((ext_vector_type(16)));
typedef __bf16 v8bf  __attribute__((ext_vector_type(8)));
typedef float  v8f   __attribute__((ext_vector_type(8)));
typedef unsigned int v4u __attribute__((ext_vector_type(4)));

union BF4 { __bf16 h[4]; unsigned long long u; };
union BF8 { __bf16 h[8]; v4u v; };

// A-fragment (16x32 bf16, MxK): lane m = lane&15, half = lane>>4.
// element i -> K = (i&7) + 8*half + 16*(i>>3)  => two 16B LDS reads.
__device__ __forceinline__ v16bf ld_frag_a(const __bf16* s, int row, int half) {
  const __bf16* p = s + row * LPITCH;
  v8bf lo = *(const v8bf*)(p + (half << 3));
  v8bf hi = *(const v8bf*)(p + 16 + (half << 3));
  v16bf o;
#pragma unroll
  for (int i = 0; i < 8; ++i) { o[i] = lo[i]; o[i + 8] = hi[i]; }
  return o;
}

// B-fragment (32x16 bf16, KxN): lane col = lane&15, element i -> K = (i&7) + 8*half + 16*(i>>3).
// LDS holds B transposed ([col][k]) so this is two 16B reads.
__device__ __forceinline__ v16bf ld_frag_b(const __bf16* s, int col, int half) {
  const __bf16* p = s + col * LPITCH;
  v8bf lo = *(const v8bf*)(p + (half << 3));
  v8bf hi = *(const v8bf*)(p + 16 + (half << 3));
  v16bf o;
#pragma unroll
  for (int i = 0; i < 8; ++i) { o[i] = lo[i]; o[i + 8] = hi[i]; }
  return o;
}

// -------------------------------------------------------------------------
// Kernel 1: AH[r] = A[r] @ H      A:[R,N,N] f32, H:[N,F] f32 -> AH:[R,N,F] f32
// Grid (N/64, R), block 256 (8 waves). Each WG: 64 rows x 256 cols (FULL F)
// => A is streamed from HBM exactly once per layer (roofline floor).
// Wave w: row-tile rt = w&3, col tiles (w>>2)*8 .. +7 -> 8 WMMAs / K-step,
// issued back-to-back with reuse_a hints (same A fragment).
// -------------------------------------------------------------------------
__global__ void rgcn_spmm_kernel(const float* __restrict__ A,
                                 const float* __restrict__ H,
                                 float* __restrict__ AH) {
  __shared__ __attribute__((aligned(16))) __bf16 sA[64 * LPITCH];    //  5 KB
  __shared__ __attribute__((aligned(16))) __bf16 sB[256 * LPITCH];   // 20 KB

  const int r        = blockIdx.y;
  const int row_base = blockIdx.x * 64;
  const int tid  = threadIdx.x;
  const int wave = tid >> 5, lane = tid & 31;
  const int half = lane >> 4, l16 = lane & 15;
  const int rt    = wave & 3;
  const int cbase = (wave >> 2) << 3;   // first of 8 col tiles (0 or 8)
  const size_t Abase = (size_t)r * NN * NN;

  v8f acc[8];
#pragma unroll
  for (int j = 0; j < 8; ++j) acc[j] = (v8f){0.f,0.f,0.f,0.f,0.f,0.f,0.f,0.f};

  for (int kb = 0; kb < NN; kb += 32) {
    // ---- Stage A tile 64x32 (fp32 -> bf16, row-major), 2 float4 / thread
#pragma unroll
    for (int it = 0; it < 2; ++it) {
      int q = tid + (it << 8);
      int arow = q >> 3, kq = (q & 7) << 2;
      const float4 va = *(const float4*)(A + Abase + (size_t)(row_base + arow) * NN + kb + kq);
      BF4 t; t.h[0] = (__bf16)va.x; t.h[1] = (__bf16)va.y;
             t.h[2] = (__bf16)va.z; t.h[3] = (__bf16)va.w;
      *(unsigned long long*)(&sA[arow * LPITCH + kq]) = t.u;
    }
    // ---- Stage H tile 32x256 transposed: thread owns one column, packed
    //      ds_store_b128 of 8 bf16 k-values (loads coalesced across threads).
    {
      const int ccol = tid;  // 0..255 == global col (F == 256)
#pragma unroll
      for (int kk = 0; kk < 32; kk += 8) {
        BF8 t;
#pragma unroll
        for (int j = 0; j < 8; ++j)
          t.h[j] = (__bf16)H[(size_t)(kb + kk + j) * FF + ccol];
        *(v4u*)(&sB[ccol * LPITCH + kk]) = t.v;
      }
    }
    // Prefetch next A tile (global_prefetch_b8)
    if (kb + 32 < NN)
      __builtin_prefetch((const void*)(A + Abase + (size_t)(row_base + (tid >> 3)) * NN
                                       + kb + 32 + ((tid & 7) << 2)), 0, 1);
    __syncthreads();

    v16bf af = ld_frag_a(sA, rt * 16 + l16, half);
    v16bf bfr[8];
#pragma unroll
    for (int j = 0; j < 8; ++j)
      bfr[j] = ld_frag_b(sB, (cbase + j) * 16 + l16, half);

    // 8 back-to-back WMMAs sharing the A fragment.
    // reuse_a must be a literal constant -> macro-expanded chain.
#define WMMA_STEP(J, RA)                                                      \
    acc[J] = __builtin_amdgcn_wmma_f32_16x16x32_bf16(false, af, false,        \
                                                     bfr[J], (short)0,        \
                                                     acc[J], RA, false)
    WMMA_STEP(0, true);
    WMMA_STEP(1, true);
    WMMA_STEP(2, true);
    WMMA_STEP(3, true);
    WMMA_STEP(4, true);
    WMMA_STEP(5, true);
    WMMA_STEP(6, true);
    WMMA_STEP(7, false);
#undef WMMA_STEP
    __syncthreads();
  }

  // C/D layout: VGPR v -> row = v + 8*half, col = lane&15
  float* outp = AH + (size_t)r * NN * FF;
#pragma unroll
  for (int j = 0; j < 8; ++j) {
    const int ocol = (cbase + j) * 16 + l16;
#pragma unroll
    for (int v = 0; v < 8; ++v) {
      int orow = row_base + rt * 16 + v + (half << 3);
      outp[(size_t)orow * FF + ocol] = acc[j][v];
    }
  }
}

// -------------------------------------------------------------------------
// Kernel 2: Hout[n,o] = sum_r c[r,n] * (AH[r] @ W[r]^T)[n,o]
// Grid (N/64, F/64), block 256.
// -------------------------------------------------------------------------
__global__ void rgcn_combine_kernel(const float* __restrict__ AH,
                                    const float* __restrict__ W,
                                    const float* __restrict__ c,
                                    float* __restrict__ Hout) {
  __shared__ __attribute__((aligned(16))) __bf16 sA[64 * LPITCH];
  __shared__ __attribute__((aligned(16))) __bf16 sB[64 * LPITCH];
  __shared__ float sC[64];

  const int row_base = blockIdx.x * 64;   // n
  const int col_base = blockIdx.y * 64;   // o
  const int tid  = threadIdx.x;
  const int wave = tid >> 5, lane = tid & 31;
  const int half = lane >> 4, l16 = lane & 15;
  const int rt   = wave & 3;
  const int ct0  = (wave >> 2) << 1;

  v8f f0 = {0.f,0.f,0.f,0.f,0.f,0.f,0.f,0.f};
  v8f f1 = {0.f,0.f,0.f,0.f,0.f,0.f,0.f,0.f};

  for (int r = 0; r < RNUM; ++r) {
    if (tid < 64) sC[tid] = c[(size_t)r * NN + row_base + tid];
    v8f acc0 = {0.f,0.f,0.f,0.f,0.f,0.f,0.f,0.f};
    v8f acc1 = {0.f,0.f,0.f,0.f,0.f,0.f,0.f,0.f};
    const float* Ar = AH + (size_t)r * NN * FF;
    const float* Wr = W  + (size_t)r * FF * FF;

    for (int kb = 0; kb < FF; kb += 32) {
#pragma unroll
      for (int it = 0; it < 2; ++it) {
        int q = tid + (it << 8);
        int arow = q >> 3, kq = (q & 7) << 2;
        const float4 va = *(const float4*)(Ar + (size_t)(row_base + arow) * FF + kb + kq);
        BF4 t; t.h[0] = (__bf16)va.x; t.h[1] = (__bf16)va.y;
               t.h[2] = (__bf16)va.z; t.h[3] = (__bf16)va.w;
        *(unsigned long long*)(&sA[arow * LPITCH + kq]) = t.u;
      }
#pragma unroll
      for (int it = 0; it < 2; ++it) {
        int q = tid + (it << 8);
        int orow = q >> 3, kq = (q & 7) << 2;
        const float4 vb = *(const float4*)(Wr + (size_t)(col_base + orow) * FF + kb + kq);
        BF4 t; t.h[0] = (__bf16)vb.x; t.h[1] = (__bf16)vb.y;
               t.h[2] = (__bf16)vb.z; t.h[3] = (__bf16)vb.w;
        *(unsigned long long*)(&sB[orow * LPITCH + kq]) = t.u;
      }
      __syncthreads();

      v16bf af = ld_frag_a(sA, rt * 16 + l16, half);
      v16bf b0 = ld_frag_b(sB, ct0 * 16 + l16, half);
      v16bf b1 = ld_frag_b(sB, (ct0 + 1) * 16 + l16, half);
      acc0 = __builtin_amdgcn_wmma_f32_16x16x32_bf16(false, af, false, b0, (short)0, acc0, true,  false);
      acc1 = __builtin_amdgcn_wmma_f32_16x16x32_bf16(false, af, false, b1, (short)0, acc1, false, false);
      __syncthreads();
    }

#pragma unroll
    for (int v = 0; v < 8; ++v) {
      float cv = sC[rt * 16 + v + (half << 3)];
      f0[v] += cv * acc0[v];
      f1[v] += cv * acc1[v];
    }
    __syncthreads();  // protect sC before next r overwrites it
  }

#pragma unroll
  for (int v = 0; v < 8; ++v) {
    int orow = row_base + rt * 16 + v + (half << 3);
    Hout[(size_t)orow * FF + col_base + ct0 * 16 + l16]       = f0[v];
    Hout[(size_t)orow * FF + col_base + (ct0 + 1) * 16 + l16] = f1[v];
  }
}

// -------------------------------------------------------------------------
// Kernel 3: out[b] = sigmoid( E1[b]^T @ M[rel[b]] @ E2[b] )
// -------------------------------------------------------------------------
__global__ void distmult_score_kernel(const float* __restrict__ H,
                                      const float* __restrict__ M,
                                      const int* __restrict__ e1,
                                      const int* __restrict__ rel,
                                      const int* __restrict__ e2,
                                      float* __restrict__ out) {
  __shared__ float sE1[FF];
  __shared__ float sP[FF];
  const int b = blockIdx.x, t = threadIdx.x;
  const int i1 = e1[b], rr = rel[b], i2 = e2[b];
  sE1[t] = H[(size_t)i1 * FF + t];
  const float e2v = H[(size_t)i2 * FF + t];
  __syncthreads();

  const float* Mr = M + (size_t)rr * FF * FF + t;
  float acc = 0.f;
#pragma unroll 4
  for (int f = 0; f < FF; ++f) acc += sE1[f] * Mr[(size_t)f * FF];
  sP[t] = acc * e2v;
  __syncthreads();
  for (int s = 128; s > 0; s >>= 1) {
    if (t < s) sP[t] += sP[t + s];
    __syncthreads();
  }
  if (t == 0) out[b] = 1.f / (1.f + __expf(-sP[0]));
}

// -------------------------------------------------------------------------
extern "C" void kernel_launch(void* const* d_in, const int* in_sizes, int n_in,
                              void* d_out, int out_size, void* d_ws, size_t ws_size,
                              hipStream_t stream) {
  const float* A    = (const float*)d_in[0];  // [R,N,N]
  const float* feat = (const float*)d_in[1];  // [N,F]
  const float* c    = (const float*)d_in[2];  // [R,N,1]
  const float* W1   = (const float*)d_in[3];  // [R,F,F]
  const float* W2   = (const float*)d_in[4];  // [R,F,F]
  const float* relm = (const float*)d_in[5];  // [R,F,F]
  const int*   e1   = (const int*)d_in[6];
  const int*   rel  = (const int*)d_in[7];
  const int*   e2   = (const int*)d_in[8];

  float* AHws = (float*)d_ws;                                                   // 32 MB
  float* Hbuf = (float*)((char*)d_ws + (size_t)RNUM * NN * FF * sizeof(float)); //  4 MB

  dim3 blk(256);
  dim3 g1(NN / 64, RNUM);
  dim3 g2(NN / 64, FF / 64);

  rgcn_spmm_kernel<<<g1, blk, 0, stream>>>(A, feat, AHws);
  rgcn_combine_kernel<<<g2, blk, 0, stream>>>(AHws, W1, c, Hbuf);
  rgcn_spmm_kernel<<<g1, blk, 0, stream>>>(A, Hbuf, AHws);
  rgcn_combine_kernel<<<g2, blk, 0, stream>>>(AHws, W2, c, Hbuf);
  distmult_score_kernel<<<dim3(BB), blk, 0, stream>>>(Hbuf, relm, e1, rel, e2, (float*)d_out);
}
